// QuantizedLinear_24721831756580
// MI455X (gfx1250) — matrix operand order repaired and checked by reading, not compiled, not run on hardware
//
#include <hip/hip_runtime.h>

// ---------------------------------------------------------------------------
// Quantized linear (int4 group-quant) as bf16 WMMA GEMM for gfx1250 (MI455X).
//   out[8192,4096] = x[8192,4096] @ dequant(packed)^T + bias
// Tiling: block = 128x128 output, K-chunk = 64, 8 waves (4x2), each wave
// computes 32x64 via 2x4 v_wmma_f32_16x16x32_bf16 accumulators (16 wmma/iter).
// Pipeline: register-staged global loads + double-buffered LDS, 1 barrier/iter.
// Staging uses native bf16 converts + b128 LDS stores to minimize VALU/DS ops.
// ---------------------------------------------------------------------------

typedef __attribute__((ext_vector_type(16))) __bf16 v16bf;
typedef __attribute__((ext_vector_type(8)))  __bf16 v8bf;
typedef __attribute__((ext_vector_type(8)))  float  v8f;

#define TOKENS 8192
#define INF    4096
#define OUTF   4096
#define BM 128
#define BN 128
#define BK 64
#define LDP 72            // 64 + 8 pad elements; 144B rows, 16B aligned, no bank conflicts
#define NIT (INF / BK)    // 64 K-iterations

__global__ __launch_bounds__(256)
void qlinear_wmma_bf16(const float* __restrict__ x,
                       const int*   __restrict__ packed,
                       const float* __restrict__ scales,
                       const float* __restrict__ offsets,
                       const float* __restrict__ bias,
                       float* __restrict__ out)
{
  __shared__ __bf16 xs[2][BM][LDP];   // x tiles, bf16
  __shared__ __bf16 ws[2][BN][LDP];   // dequantized weight tiles, bf16 [N][K]

  const int tid  = threadIdx.x;
  const int lane = tid & 31;
  const int wid  = tid >> 5;
  const int wm   = wid & 3;    // wave's M slot (4 -> 128 rows)
  const int wn   = wid >> 2;   // wave's N slot (2 -> 128 cols)
  const int lh   = lane >> 4;  // lane half (0/1)
  const int l15  = lane & 15;

  const int n0 = blockIdx.x * BN;
  const int m0 = blockIdx.y * BM;

  v8f acc[2][4];
#pragma unroll
  for (int i = 0; i < 2; ++i)
#pragma unroll
    for (int j = 0; j < 4; ++j)
      acc[i][j] = (v8f){0.f, 0.f, 0.f, 0.f, 0.f, 0.f, 0.f, 0.f};

  // Weight-loader mapping: 2 threads per output row; each handles 16 packed
  // int32 entries (= 32 weights = half of the 64-wide K chunk).
  const int wrow  = tid >> 1;   // 0..127 (local N row)
  const int whalf = tid & 1;    // 0/1

  // ---- register staging for the next tile ----
  float4 xr[8];
  int4   wr[4];
  float  wscale = 0.f, woff = 0.f;

  auto gload = [&](int it) {
    const int kk = it * BK;
    // x: each thread owns 4 slots of 8 contiguous floats (two adjacent b128s)
#pragma unroll
    for (int i = 0; i < 4; ++i) {
      int idx = tid + i * 256;        // 0..1023 slots of 8 elements
      int row = idx >> 3;             // 8 slots per 64-wide row
      int c8  = (idx & 7) << 3;       // 0..56 step 8
      const float* gp = x + (size_t)(m0 + row) * INF + kk + c8;
      xr[2 * i]     = *(const float4*)gp;
      xr[2 * i + 1] = *(const float4*)(gp + 4);
    }
    const int* pb = packed + (size_t)(n0 + wrow) * (INF / 2) + (kk >> 1) + whalf * 16;
#pragma unroll
    for (int i = 0; i < 4; ++i)
      wr[i] = *(const int4*)(pb + i * 4);
    // group index: flat (n*4096 + k)/128 = n*32 + k/128; constant over 64-chunk
    const int grp = (n0 + wrow) * (INF / 128) + (kk >> 7);
    wscale = scales[grp];
    woff   = offsets[grp];
  };

  auto stage = [&](int buf) {
    // x: native bf16 converts, one ds_store_b128 per 8 elements
#pragma unroll
    for (int i = 0; i < 4; ++i) {
      int idx = tid + i * 256;
      int row = idx >> 3;
      int c8  = (idx & 7) << 3;
      float4 a = xr[2 * i], b = xr[2 * i + 1];
      v8bf v;
      v[0] = (__bf16)a.x; v[1] = (__bf16)a.y; v[2] = (__bf16)a.z; v[3] = (__bf16)a.w;
      v[4] = (__bf16)b.x; v[5] = (__bf16)b.y; v[6] = (__bf16)b.z; v[7] = (__bf16)b.w;
      *(v8bf*)&xs[buf][row][c8] = v;
    }
    // W: dequantize 8 weights per int4 word-pair, one ds_store_b128 each
#pragma unroll
    for (int i = 0; i < 4; ++i) {
      int q4[4] = {wr[i].x, wr[i].y, wr[i].z, wr[i].w};
      v8bf v;
#pragma unroll
      for (int j = 0; j < 4; ++j) {
        int q = q4[j];
        v[2 * j]     = (__bf16)((float)(q & 15) * wscale + woff);
        v[2 * j + 1] = (__bf16)((float)((q >> 4) & 15) * wscale + woff);
      }
      *(v8bf*)&ws[buf][wrow][whalf * 32 + i * 8] = v;
    }
  };

  auto compute = [&](int buf) {
#pragma unroll
    for (int ks = 0; ks < BK; ks += 32) {
      // A fragments (ISA 16-bit A layout: lane half -> K runs +0/+8 and +16/+24)
      v16bf afrag[2];
#pragma unroll
      for (int mt = 0; mt < 2; ++mt) {
        int row = wm * 32 + mt * 16 + l15;
        v8bf lo = *(const v8bf*)&xs[buf][row][ks + lh * 8];
        v8bf hi = *(const v8bf*)&xs[buf][row][ks + lh * 8 + 16];
#pragma unroll
        for (int i = 0; i < 8; ++i) { afrag[mt][i] = lo[i]; afrag[mt][i + 8] = hi[i]; }
      }
      // B fragments (same K layout as A: lane half -> K runs +0/+8 and +16/+24) + WMMA
#pragma unroll
      for (int nt = 0; nt < 4; ++nt) {
        int row = wn * 64 + nt * 16 + l15;
        v8bf b0 = *(const v8bf*)&ws[buf][row][ks + lh * 8];
        v8bf b1 = *(const v8bf*)&ws[buf][row][ks + lh * 8 + 16];
        v16bf bfrag;
#pragma unroll
        for (int i = 0; i < 8; ++i) { bfrag[i] = b0[i]; bfrag[i + 8] = b1[i]; }
#pragma unroll
        for (int mt = 0; mt < 2; ++mt) {
          acc[mt][nt] = __builtin_amdgcn_wmma_f32_16x16x32_bf16(
              /*neg_a=*/false, afrag[mt],
              /*neg_b=*/false, bfrag,
              /*c_mod=*/(short)0, acc[mt][nt],
              /*reuse_a=*/false, /*reuse_b=*/false);
        }
      }
    }
  };

  // ---- software pipeline: prologue fills buffer 0 ----
  gload(0);
  stage(0);
  __syncthreads();

  for (int it = 0; it < NIT; ++it) {
    const int cur = it & 1;
    if (it + 1 < NIT) gload(it + 1);   // issue global loads (latency hidden by WMMAs)
    compute(cur);                       // 16 x v_wmma from LDS buffer `cur`
    if (it + 1 < NIT) stage(cur ^ 1);   // convert + ds_store into the other buffer
    __syncthreads();                    // single barrier per iteration
  }

  // ---- epilogue: C/D layout (M = r + 8*lh, N = l15), add bias, store ----
#pragma unroll
  for (int mt = 0; mt < 2; ++mt) {
#pragma unroll
    for (int nt = 0; nt < 4; ++nt) {
      int col = n0 + wn * 64 + nt * 16 + l15;
      float bv = bias[col];
#pragma unroll
      for (int r = 0; r < 8; ++r) {
        int row = m0 + wm * 32 + mt * 16 + lh * 8 + r;
        out[(size_t)row * OUTF + col] = acc[mt][nt][r] + bv;
      }
    }
  }
}

extern "C" void kernel_launch(void* const* d_in, const int* in_sizes, int n_in,
                              void* d_out, int out_size, void* d_ws, size_t ws_size,
                              hipStream_t stream) {
  const float* x       = (const float*)d_in[0];
  const int*   packed  = (const int*)d_in[1];
  const float* scales  = (const float*)d_in[2];
  const float* offsets = (const float*)d_in[3];
  const float* bias    = (const float*)d_in[4];
  float*       out     = (float*)d_out;

  dim3 grid(OUTF / BN, TOKENS / BM);  // (32, 64)
  dim3 block(256);
  qlinear_wmma_bf16<<<grid, block, 0, stream>>>(x, packed, scales, offsets, bias, out);
}
